// EvolveGCNO_13907104104627
// MI455X (gfx1250) — matrix operand order repaired and checked by reading, not compiled, not run on hardware
//
#include <hip/hip_runtime.h>
#include <cmath>

// ---------------------------------------------------------------------------
// EvolveGCN-O, algebraically collapsed:
//   W0_5 = LSTM^6(W00), W1_5 = LSTM^6(W01)       (weight evolution, data-free)
//   y1   = gcn_conv(feats[5], src5, dst5, W0_5)
//   y2   = gcn_conv(y1,       src5, dst5, W1_5)
//   out  = relu(y2 @ mlp_w1 + b1) @ mlp_w2 + b2
// Only t=5 matters because the scan carry is W alone and the MLP reads h[-1].
// Heavy GEMMs use v_wmma_f32_16x16x32_bf16 (f32 accumulate).
// ---------------------------------------------------------------------------

typedef __bf16 bf16;
typedef __attribute__((ext_vector_type(8)))  float f32x8;
typedef __attribute__((ext_vector_type(8)))  bf16  bf16x8;
typedef __attribute__((ext_vector_type(16))) bf16  bf16x16;

#define RRELU_SLOPE 0.22916666666666666f

static constexpr int T_  = 6;
static constexpr int N_  = 50000;
static constexpr int E_  = 800000;
static constexpr int F_  = 128;
static constexpr int H_  = 128;
static constexpr int CH_ = 307;
static constexpr int CHP = 320;   // padded (20 tiles of 16)
static constexpr int NC_ = 2;

__device__ __forceinline__ float sigmoidf_(float x) { return 1.0f / (1.0f + __expf(-x)); }

// ---- generic zero ---------------------------------------------------------
__global__ void k_zero(float* __restrict__ p, long long n) {
  long long i = (long long)blockIdx.x * blockDim.x + threadIdx.x;
  if (i < n) p[i] = 0.0f;
}

// ---- LSTM cell step on the 128x128 weight matrix (rows = batch) -----------
// gates = W @ wih^T + (bih + bhh); i,f,g,o split; c = sig(i)*tanh(g);
// Wout = sig(o)*tanh(c).  One block per row, 512 threads (one per gate col).
__global__ void k_lstm_step(const float* __restrict__ Win,
                            const float* __restrict__ wih,
                            const float* __restrict__ bih,
                            const float* __restrict__ bhh,
                            float* __restrict__ Wout) {
  __shared__ float sW[H_];
  __shared__ float sg[4 * H_];
  const int row = blockIdx.x;
  const int col = threadIdx.x;           // 0..511
  if (col < H_) sW[col] = Win[row * H_ + col];
  __syncthreads();
  float acc = bih[col] + bhh[col];
  const float* wr = wih + (long long)col * H_;
  #pragma unroll 4
  for (int k = 0; k < H_; ++k) acc = fmaf(sW[k], wr[k], acc);
  sg[col] = acc;
  __syncthreads();
  if (col < H_) {
    float gi = sg[col];
    float gg = sg[2 * H_ + col];
    float go = sg[3 * H_ + col];
    float c  = sigmoidf_(gi) * tanhf(gg);
    Wout[row * H_ + col] = sigmoidf_(go) * tanhf(c);
  }
}

// ---- pack a (K x Ncols) f32 weight into WMMA B-fragments (bf16) -----------
// Fragment (tile,ks) holds the 32x16 sub-matrix in ISA lane layout:
//   elem s=2v+h: K = ks*32 + 8*(lane>>4) + s (s<8), ks*32 + 16 + 8*(lane>>4) + (s-8) (s>=8)
//   col = tile*16 + (lane&15); per-lane 16 contiguous bf16 (32B).
__global__ void k_pack_b(const float* __restrict__ W, int Krows, int Ncols,
                         bf16* __restrict__ out) {
  const int frag = blockIdx.x;          // tile*4 + ks
  const int ks   = frag & 3;
  const int tile = frag >> 2;
  const int idx  = threadIdx.x;         // 0..511
  const int lane = idx & 31;
  const int v    = (idx >> 5) & 7;
  const int hb   = (idx >> 8) & 1;
  const int kk   = ((lane >> 4) << 3) + 2 * v + hb + ((v >> 2) << 3);
  const int K    = ks * 32 + kk;
  const int col  = tile * 16 + (lane & 15);
  float val = (K < Krows && col < Ncols) ? W[(long long)K * Ncols + col] : 0.0f;
  out[(long long)frag * 512 + lane * 16 + 2 * v + hb] = (bf16)val;
}

// ---- degrees for snapshot t=5 ---------------------------------------------
__global__ void k_degrees(const int* __restrict__ src, const int* __restrict__ dst,
                          float* __restrict__ degOut, float* __restrict__ degIn, int E) {
  int e = blockIdx.x * blockDim.x + threadIdx.x;
  if (e < E) {
    __hip_atomic_fetch_add(&degOut[src[e]], 1.0f, __ATOMIC_RELAXED, __HIP_MEMORY_SCOPE_AGENT);
    __hip_atomic_fetch_add(&degIn[dst[e]],  1.0f, __ATOMIC_RELAXED, __HIP_MEMORY_SCOPE_AGENT);
  }
}

// ---- Xbf = bf16(feats5 * rsqrt(max(degOut,1))) ----------------------------
__global__ void k_scale_convert(const float* __restrict__ X,
                                const float* __restrict__ degOut,
                                bf16* __restrict__ Xbf, int Nn) {
  long long i = (long long)blockIdx.x * blockDim.x + threadIdx.x;
  if (i < (long long)Nn * H_) {
    int n = (int)(i >> 7);
    float rs = rsqrtf(fmaxf(degOut[n], 1.0f));
    Xbf[i] = (bf16)(X[i] * rs);
  }
}

// ---- WMMA GEMM: C[M x Ncols] = A[M x 128](bf16) @ Bpacked -----------------
// 256 threads = 8 waves; wave w owns rows m0..m0+15; NTILES n-tiles of 16.
template <int NTILES>
__global__ void k_gemm_wmma(const bf16* __restrict__ A, const bf16* __restrict__ Bp,
                            float* __restrict__ C, int M, int Ncols, int ldC,
                            int tileOff, const float* __restrict__ bias, int doRelu) {
  const int lane = threadIdx.x & 31;
  const int wave = threadIdx.x >> 5;
  const int m0   = blockIdx.x * 128 + wave * 16;
  int arow = m0 + (lane & 15);
  if (arow > M - 1) arow = M - 1;                    // clamp loads; stores guarded
  const bf16* Arow = A + (long long)arow * H_;
  const int koff = (lane >> 4) * 8;                  // lanes 16-31: K runs 8..15 / 24..31

  f32x8 zero = {0.f, 0.f, 0.f, 0.f, 0.f, 0.f, 0.f, 0.f};
  f32x8 acc[NTILES];
  #pragma unroll
  for (int n = 0; n < NTILES; ++n) acc[n] = zero;

  #pragma unroll
  for (int ks = 0; ks < 4; ++ks) {
    bf16x8 lo = *(const bf16x8*)(Arow + ks * 32 + koff);
    bf16x8 hi = *(const bf16x8*)(Arow + ks * 32 + koff + 16);
    bf16x16 a = __builtin_shufflevector(lo, hi, 0, 1, 2, 3, 4, 5, 6, 7,
                                                8, 9, 10, 11, 12, 13, 14, 15);
    #pragma unroll
    for (int n = 0; n < NTILES; ++n) {
      const bf16x16 b = *(const bf16x16*)(Bp + ((long long)(tileOff + n) * 4 + ks) * 512 + lane * 16);
      acc[n] = __builtin_amdgcn_wmma_f32_16x16x32_bf16(
                   false, a, false, b, (short)0, acc[n], false, false);
    }
  }

  const int rbase = m0 + ((lane >> 4) * 8);
  const int cl = lane & 15;
  #pragma unroll
  for (int n = 0; n < NTILES; ++n) {
    int col = (tileOff + n) * 16 + cl;
    if (col >= Ncols) continue;
    float bv = bias ? bias[col] : 0.0f;
    #pragma unroll
    for (int r = 0; r < 8; ++r) {
      int row = rbase + r;
      if (row < M) {
        float val = acc[n][r] + bv;
        if (doRelu) val = fmaxf(val, 0.0f);
        C[(long long)row * ldC + col] = val;
      }
    }
  }
}

// ---- edge scatter: agg[dst] += h[src] (f32 atomics, 32 lanes x 4 floats) ---
__global__ void k_scatter_add(const float* __restrict__ h,
                              const int* __restrict__ src, const int* __restrict__ dst,
                              float* __restrict__ agg, int E) {
  const int e    = blockIdx.x * 8 + (threadIdx.x >> 5);
  const int lane = threadIdx.x & 31;
  if (e < E) {
    const int s = src[e], d = dst[e];
    const float4 v = *(const float4*)(h + (long long)s * H_ + lane * 4);
    float* p = agg + (long long)d * H_ + lane * 4;
    __hip_atomic_fetch_add(p + 0, v.x, __ATOMIC_RELAXED, __HIP_MEMORY_SCOPE_AGENT);
    __hip_atomic_fetch_add(p + 1, v.y, __ATOMIC_RELAXED, __HIP_MEMORY_SCOPE_AGENT);
    __hip_atomic_fetch_add(p + 2, v.z, __ATOMIC_RELAXED, __HIP_MEMORY_SCOPE_AGENT);
    __hip_atomic_fetch_add(p + 3, v.w, __ATOMIC_RELAXED, __HIP_MEMORY_SCOPE_AGENT);
  }
}

// ---- epilogue: y = rrelu(agg * rsqrt(degIn)); optionally pre-scale for next
// conv (multiply by rsqrt(degOut)) and emit bf16 A-matrix for next GEMM -----
__global__ void k_finalize(const float* __restrict__ agg,
                           const float* __restrict__ degIn,
                           const float* __restrict__ degOut,
                           bf16* __restrict__ outBf, int Nn, int scaleOut) {
  long long i = (long long)blockIdx.x * blockDim.x + threadIdx.x;
  if (i < (long long)Nn * H_) {
    int n = (int)(i >> 7);
    float v = agg[i] * rsqrtf(fmaxf(degIn[n], 1.0f));
    v = (v >= 0.0f) ? v : v * RRELU_SLOPE;
    if (scaleOut) v *= rsqrtf(fmaxf(degOut[n], 1.0f));
    outBf[i] = (bf16)v;
  }
}

// ---- final tiny GEMM: out[n,0..1] = relu'd hidden[n,:307] @ w2 + b2 -------
__global__ void k_mlp2(const float* __restrict__ hidden, const float* __restrict__ w2,
                       const float* __restrict__ b2, float* __restrict__ out, int M) {
  int n = blockIdx.x * blockDim.x + threadIdx.x;
  if (n >= M) return;
  const float* hr = hidden + (long long)n * CHP;
  float a0 = b2[0], a1 = b2[1];
  for (int j = 0; j < CH_; ++j) {
    float hv = hr[j];
    a0 = fmaf(hv, w2[j * 2 + 0], a0);
    a1 = fmaf(hv, w2[j * 2 + 1], a1);
  }
  out[n * 2 + 0] = a0;
  out[n * 2 + 1] = a1;
}

// ---------------------------------------------------------------------------
extern "C" void kernel_launch(void* const* d_in, const int* in_sizes, int n_in,
                              void* d_out, int out_size, void* d_ws, size_t ws_size,
                              hipStream_t stream) {
  const float* feats  = (const float*)d_in[0];
  const int*   src    = (const int*)d_in[1];
  const int*   dst    = (const int*)d_in[2];
  const float* W00    = (const float*)d_in[3];
  const float* W01    = (const float*)d_in[4];
  const float* l0_wih = (const float*)d_in[5];
  const float* l0_bih = (const float*)d_in[6];
  const float* l0_bhh = (const float*)d_in[7];
  const float* l1_wih = (const float*)d_in[8];
  const float* l1_bih = (const float*)d_in[9];
  const float* l1_bhh = (const float*)d_in[10];
  const float* mlp_w1 = (const float*)d_in[11];
  const float* mlp_b1 = (const float*)d_in[12];
  const float* mlp_w2 = (const float*)d_in[13];
  const float* mlp_b2 = (const float*)d_in[14];
  float* out = (float*)d_out;

  const int* src5 = src + 5 * E_;
  const int* dst5 = dst + 5 * E_;
  const float* feats5 = feats + (long long)5 * N_ * F_;

  // --- workspace carve-out (256B aligned) ---
  char* base = (char*)d_ws;
  size_t off = 0;
  auto carve = [&](size_t bytes) -> char* {
    char* p = base + off;
    off = (off + bytes + 255) & ~(size_t)255;
    return p;
  };
  float* Wa      = (float*)carve(H_ * H_ * 4);
  float* Wb      = (float*)carve(H_ * H_ * 4);
  bf16*  Bp0     = (bf16*)carve(8  * 4 * 512 * 2);
  bf16*  Bp1     = (bf16*)carve(8  * 4 * 512 * 2);
  bf16*  BpM     = (bf16*)carve(20 * 4 * 512 * 2);
  float* degOut  = (float*)carve((size_t)N_ * 4);
  float* degIn   = (float*)carve((size_t)N_ * 4);
  bf16*  Xbf     = (bf16*)carve((size_t)N_ * H_ * 2);
  float* hbuf    = (float*)carve((size_t)N_ * H_ * 4);
  float* agg     = (float*)carve((size_t)N_ * H_ * 4);
  float* hidden  = (float*)carve((size_t)N_ * CHP * 4);

  const long long NH = (long long)N_ * H_;
  const int gNH  = (int)((NH + 255) / 256);
  const int gM   = (N_ + 127) / 128;   // 391 blocks, 8 waves x 16 rows each

  // 1) evolve W0 = LSTM^6(W00), pack B-fragments
  {
    const float* cur = W00;
    for (int s = 0; s < T_; ++s) {
      float* o = (s & 1) ? Wb : Wa;
      k_lstm_step<<<H_, 4 * H_, 0, stream>>>(cur, l0_wih, l0_bih, l0_bhh, o);
      cur = o;
    }
    k_pack_b<<<8 * 4, 512, 0, stream>>>(cur, H_, H_, Bp0);
  }
  // 2) evolve W1 = LSTM^6(W01), pack
  {
    const float* cur = W01;
    for (int s = 0; s < T_; ++s) {
      float* o = (s & 1) ? Wb : Wa;
      k_lstm_step<<<H_, 4 * H_, 0, stream>>>(cur, l1_wih, l1_bih, l1_bhh, o);
      cur = o;
    }
    k_pack_b<<<8 * 4, 512, 0, stream>>>(cur, H_, H_, Bp1);
  }
  // 3) pack mlp_w1 (128 x 307 -> 20 padded tiles)
  k_pack_b<<<20 * 4, 512, 0, stream>>>(mlp_w1, H_, CH_, BpM);

  // 4) degrees for t=5
  k_zero<<<(2 * N_ + 255) / 256, 256, 0, stream>>>(degOut, 2LL * N_); // degOut+degIn adjacent-ish
  k_zero<<<(N_ + 255) / 256, 256, 0, stream>>>(degIn, N_);           // (degIn alignment gap safe)
  k_degrees<<<(E_ + 255) / 256, 256, 0, stream>>>(src5, dst5, degOut, degIn, E_);

  // 5) conv 1: project, scatter, normalize+rrelu (+pre-scale for conv 2)
  k_scale_convert<<<gNH, 256, 0, stream>>>(feats5, degOut, Xbf, N_);
  k_gemm_wmma<8><<<gM, 256, 0, stream>>>(Xbf, Bp0, hbuf, N_, H_, H_, 0, nullptr, 0);
  k_zero<<<gNH, 256, 0, stream>>>(agg, NH);
  k_scatter_add<<<(E_ + 7) / 8, 256, 0, stream>>>(hbuf, src5, dst5, agg, E_);
  k_finalize<<<gNH, 256, 0, stream>>>(agg, degIn, degOut, Xbf, N_, /*scaleOut=*/1);

  // 6) conv 2
  k_gemm_wmma<8><<<gM, 256, 0, stream>>>(Xbf, Bp1, hbuf, N_, H_, H_, 0, nullptr, 0);
  k_zero<<<gNH, 256, 0, stream>>>(agg, NH);
  k_scatter_add<<<(E_ + 7) / 8, 256, 0, stream>>>(hbuf, src5, dst5, agg, E_);
  k_finalize<<<gNH, 256, 0, stream>>>(agg, degIn, degOut, Xbf, N_, /*scaleOut=*/0);

  // 7) MLP: hidden = relu(last @ w1 + b1) via WMMA (two 10-tile halves)
  k_gemm_wmma<10><<<gM, 256, 0, stream>>>(Xbf, BpM, hidden, N_, CH_, CHP, 0,  mlp_b1, 1);
  k_gemm_wmma<10><<<gM, 256, 0, stream>>>(Xbf, BpM, hidden, N_, CH_, CHP, 10, mlp_b1, 1);

  // 8) out = hidden @ w2 + b2
  k_mlp2<<<(N_ + 255) / 256, 256, 0, stream>>>(hidden, mlp_w2, mlp_b2, out, N_);
}
